// ConvolutionLayer_86595130622252
// MI455X (gfx1250) — matrix operand order repaired and checked by reading, not compiled, not run on hardware
//
#include <hip/hip_runtime.h>
#include <hip/hip_bf16.h>
#include <math.h>

// CDNA5 (gfx1250) fused tensor-field conv layer, round 2.
// 5 tiny-MLP evals (64->64->32) over 262144 rows via v_wmma_f32_16x16x32_f16.
// - one wave == one 16-row M tile, 4 tiles per wave (amortized weight staging)
// - filter-paired evaluation (f0 on x0&x1, f11 on x0&x1) shares B fragments
// - non-temporal loads/stores for the streamed 0.5 GB of traffic
// - global_prefetch for the next tile's activations

typedef __attribute__((ext_vector_type(16))) _Float16 v16h;
typedef __attribute__((ext_vector_type(8)))  _Float16 v8h;
typedef __attribute__((ext_vector_type(8)))  float    v8f;
typedef __attribute__((ext_vector_type(4)))  float    v4f;

#define NPT   512
#define DIM   64
#define HID   64
#define OUTD  32
#define EPSF  1e-8f
#define TPW   4          // tiles per wave

__device__ __forceinline__ v8f wmma16(v16h a, v16h b, v8f c) {
  // (neg_a, A, neg_b, B, c_mod, C, reuse_a, reuse_b)
  return __builtin_amdgcn_wmma_f32_16x16x32_f16(false, a, false, b, (short)0, c,
                                                false, false);
}

// Build the 16-bit A-fragment (16x32, MxK) from a global f32 row-major tile.
// lr = lane&15 -> row M=lr; hl = lane>>4;
//   elements 0..7  = K kbase + 16*hl + 0..7
//   elements 8..15 = K kbase + 16*hl + 8 + 0..7
__device__ __forceinline__ v16h load_xA(const float* __restrict__ rowbase,
                                        int ks, int hl) {
  v16h a;
#pragma unroll
  for (int g = 0; g < 2; ++g) {
    const float* p = rowbase + ks * 32 + 16 * hl + g * 8;
    v4f f0 = __builtin_nontemporal_load((const v4f*)(p));
    v4f f1 = __builtin_nontemporal_load((const v4f*)(p + 4));
#pragma unroll
    for (int i = 0; i < 4; ++i) {
      a[g * 8 + i]     = (_Float16)f0[i];
      a[g * 8 + 4 + i] = (_Float16)f1[i];
    }
  }
  return a;
}

// Re-layout one 16x64 f16 hidden tile (row-major in LDS) into two A fragments.
__device__ __forceinline__ void hid_to_A(const _Float16* __restrict__ hid,
                                         int lr, int hl, v16h& hA0, v16h& hA1) {
#pragma unroll
  for (int ks = 0; ks < 2; ++ks) {
    const _Float16* pr = hid + lr * HID + ks * 32 + 16 * hl;
    v8h lo = *(const v8h*)(pr);       // K kbase+16*hl+0..7
    v8h hi = *(const v8h*)(pr + 8);   // K kbase+16*hl+8..15
    v16h a;
#pragma unroll
    for (int i = 0; i < 8; ++i) { a[i] = lo[i]; a[i + 8] = hi[i]; }
    if (ks == 0) hA0 = a; else hA1 = a;
  }
}

struct Rad { v8f r0, r1; };   // two 16-col C fragments: cols 0..15, 16..31

// Paired MLP: same weights, two inputs (A/B). Shares B fragments + biases.
__device__ __forceinline__ void mlp_rad_pair(
    v16h aAK0, v16h aAK1, v16h aBK0, v16h aBK1,
    const _Float16* __restrict__ w1T, const _Float16* __restrict__ w2T,
    const float* __restrict__ b1, const float* __restrict__ b2,
    _Float16* __restrict__ hid, int lr, int hl, Rad& outA, Rad& outB) {
  v8f hB[4];  // x_B hidden tiles kept in registers while x_A uses LDS slice
  // ---- GEMM1: hidden = relu(x@W1 + b1), both inputs per B-frag load ----
#pragma unroll
  for (int nt = 0; nt < 4; ++nt) {
    float bias = b1[nt * 16 + lr];
    v8f accA, accB;
#pragma unroll
    for (int i = 0; i < 8; ++i) { accA[i] = bias; accB[i] = bias; }
    const _Float16* pb = w1T + (nt * 16 + lr) * DIM + 16 * hl;
    v16h b0  = *(const v16h*)(pb);        // K = 0..31 slice
    v16h b1f = *(const v16h*)(pb + 32);   // K = 32..63 slice
    accA = wmma16(aAK0, b0, accA);
    accA = wmma16(aAK1, b1f, accA);
    accB = wmma16(aBK0, b0, accB);
    accB = wmma16(aBK1, b1f, accB);
#pragma unroll
    for (int i = 0; i < 8; ++i) {
      accA[i] = accA[i] > 0.0f ? accA[i] : 0.0f;
      accB[i] = accB[i] > 0.0f ? accB[i] : 0.0f;
    }
#pragma unroll
    for (int r = 0; r < 8; ++r)
      hid[(r + 8 * hl) * HID + nt * 16 + lr] = (_Float16)accA[r];
    hB[nt] = accB;
  }
  __asm volatile("s_wait_dscnt 0" ::: "memory");
  v16h hA0, hA1;
  hid_to_A(hid, lr, hl, hA0, hA1);          // x_A hidden -> A frags
  // overwrite LDS slice with x_B hidden (in-order LDS: stores after loads)
#pragma unroll
  for (int nt = 0; nt < 4; ++nt)
#pragma unroll
    for (int r = 0; r < 8; ++r)
      hid[(r + 8 * hl) * HID + nt * 16 + lr] = (_Float16)hB[nt][r];
  __asm volatile("s_wait_dscnt 0" ::: "memory");
  v16h hB0, hB1;
  hid_to_A(hid, lr, hl, hB0, hB1);
  // ---- GEMM2: rad = hidden@W2 + b2, both inputs per B-frag load ----
#pragma unroll
  for (int nt = 0; nt < 2; ++nt) {
    float bias = b2[nt * 16 + lr];
    v8f accA, accB;
#pragma unroll
    for (int i = 0; i < 8; ++i) { accA[i] = bias; accB[i] = bias; }
    const _Float16* pb = w2T + (nt * 16 + lr) * HID + 16 * hl;
    v16h b0  = *(const v16h*)(pb);
    v16h b1f = *(const v16h*)(pb + 32);
    accA = wmma16(hA0, b0, accA);
    accA = wmma16(hA1, b1f, accA);
    accB = wmma16(hB0, b0, accB);
    accB = wmma16(hB1, b1f, accB);
    if (nt == 0) { outA.r0 = accA; outB.r0 = accB; }
    else         { outA.r1 = accA; outB.r1 = accB; }
  }
}

// Single-input MLP (for f10 on x1).
__device__ __forceinline__ Rad mlp_rad(v16h aK0, v16h aK1,
                                       const _Float16* __restrict__ w1T,
                                       const _Float16* __restrict__ w2T,
                                       const float* __restrict__ b1,
                                       const float* __restrict__ b2,
                                       _Float16* __restrict__ hid,
                                       int lr, int hl) {
#pragma unroll
  for (int nt = 0; nt < 4; ++nt) {
    float bias = b1[nt * 16 + lr];
    v8f acc;
#pragma unroll
    for (int i = 0; i < 8; ++i) acc[i] = bias;
    const _Float16* pb = w1T + (nt * 16 + lr) * DIM + 16 * hl;
    v16h b0  = *(const v16h*)(pb);
    v16h b1f = *(const v16h*)(pb + 32);
    acc = wmma16(aK0, b0, acc);
    acc = wmma16(aK1, b1f, acc);
#pragma unroll
    for (int i = 0; i < 8; ++i) acc[i] = acc[i] > 0.0f ? acc[i] : 0.0f;
#pragma unroll
    for (int r = 0; r < 8; ++r)
      hid[(r + 8 * hl) * HID + nt * 16 + lr] = (_Float16)acc[r];
  }
  __asm volatile("s_wait_dscnt 0" ::: "memory");
  v16h hA0, hA1;
  hid_to_A(hid, lr, hl, hA0, hA1);
  Rad out;
#pragma unroll
  for (int nt = 0; nt < 2; ++nt) {
    float bias = b2[nt * 16 + lr];
    v8f acc;
#pragma unroll
    for (int i = 0; i < 8; ++i) acc[i] = bias;
    const _Float16* pb = w2T + (nt * 16 + lr) * HID + 16 * hl;
    v16h b0  = *(const v16h*)(pb);
    v16h b1f = *(const v16h*)(pb + 32);
    acc = wmma16(hA0, b0, acc);
    acc = wmma16(hA1, b1f, acc);
    if (nt == 0) out.r0 = acc; else out.r1 = acc;
  }
  return out;
}

__device__ __forceinline__ void store_out0(float* __restrict__ base, v8f rad,
                                           size_t row0, int nt, int lr, int hl) {
#pragma unroll
  for (int r = 0; r < 8; ++r) {
    size_t m = row0 + (size_t)(r + 8 * hl);
    __builtin_nontemporal_store(rad[r], base + m * OUTD + (size_t)(nt * 16 + lr));
  }
}

__device__ __forceinline__ void store_out1(float* __restrict__ base, v8f rad,
                                           const float* ux, const float* uy,
                                           const float* uz, size_t row0, int nt,
                                           int lr, int hl) {
#pragma unroll
  for (int r = 0; r < 8; ++r) {
    size_t m = row0 + (size_t)(r + 8 * hl);
    float* p = base + (m * OUTD + (size_t)(nt * 16 + lr)) * 3;
    __builtin_nontemporal_store(ux[r] * rad[r], p + 0);
    __builtin_nontemporal_store(uy[r] * rad[r], p + 1);
    __builtin_nontemporal_store(uz[r] * rad[r], p + 2);
  }
}

__global__ __launch_bounds__(256) void tfn_conv_kernel(
    const float* __restrict__ x0, const float* __restrict__ x1,
    const float* __restrict__ rij,
    const float* __restrict__ f0_w1, const float* __restrict__ f0_b1,
    const float* __restrict__ f0_w2, const float* __restrict__ f0_b2,
    const float* __restrict__ f10_w1, const float* __restrict__ f10_b1,
    const float* __restrict__ f10_w2, const float* __restrict__ f10_b2,
    const float* __restrict__ f11_w1, const float* __restrict__ f11_b1,
    const float* __restrict__ f11_w2, const float* __restrict__ f11_b2,
    float* __restrict__ out) {
  __shared__ __align__(32) _Float16 sW1T[3][HID * DIM];   // transposed [n][k]
  __shared__ __align__(32) _Float16 sW2T[3][OUTD * HID];  // transposed [o][k]
  __shared__ __align__(16) float sB1[3][HID];
  __shared__ __align__(16) float sB2[3][OUTD];
  __shared__ __align__(32) _Float16 sHid[8][16 * HID];    // per-wave hidden
  __shared__ __align__(16) float sU[8][16][4];            // per-wave unit vecs

  const int tid = threadIdx.x;

  // ---- stage weights once per block (f32 -> f16, transposed) ----
  const float* w1s[3] = { f0_w1, f10_w1, f11_w1 };
  const float* w2s[3] = { f0_w2, f10_w2, f11_w2 };
#pragma unroll
  for (int f = 0; f < 3; ++f) {
    for (int i = tid; i < DIM * HID; i += 256) {
      int k = i >> 6, n = i & 63;                      // w1[k][n]
      sW1T[f][n * DIM + k] = (_Float16)w1s[f][i];
    }
    for (int i = tid; i < HID * OUTD; i += 256) {
      int k = i >> 5, o = i & 31;                      // w2[k][o]
      sW2T[f][o * HID + k] = (_Float16)w2s[f][i];
    }
  }
  if (tid < HID) {
    sB1[0][tid] = f0_b1[tid]; sB1[1][tid] = f10_b1[tid]; sB1[2][tid] = f11_b1[tid];
  }
  if (tid < OUTD) {
    sB2[0][tid] = f0_b2[tid]; sB2[1][tid] = f10_b2[tid]; sB2[2][tid] = f11_b2[tid];
  }
  __syncthreads();

  const int wv = tid >> 5;
  const int lane = tid & 31;
  const int hl = lane >> 4;   // half-wave selector
  const int lr = lane & 15;   // row / column within 16

  const size_t S0 = (size_t)NPT * NPT * OUTD;  // 8388608
  float* out0a = out;
  float* out0b = out + S0;
  float* out1a = out + 2 * S0;
  float* out1b = out + 5 * S0;
  float* out1c = out + 8 * S0;
  _Float16* hid = sHid[wv];
  float* uRow = &sU[wv][0][0];

  const int tile0 = blockIdx.x * (8 * TPW) + wv * TPW;

  for (int it = 0; it < TPW; ++it) {
    const int tile = tile0 + it;
    const size_t row0 = (size_t)tile * 16;

    const float* x0row = x0 + (row0 + (size_t)lr) * DIM;
    const float* x1row = x1 + (row0 + (size_t)lr) * DIM;
    if (it + 1 < TPW) {  // prefetch next tile's activations into L2
      __builtin_prefetch(x0row + 16 * DIM, 0, 1);
      __builtin_prefetch(x1row + 16 * DIM, 0, 1);
    }

    // ---- load x0/x1 A fragments (f32 -> f16, non-temporal) ----
    v16h a0K0 = load_xA(x0row, 0, hl), a0K1 = load_xA(x0row, 1, hl);
    v16h a1K0 = load_xA(x1row, 0, hl), a1K1 = load_xA(x1row, 1, hl);

    // ---- masked unit vectors per row (lanes 0..15), broadcast via LDS ----
    if (lane < 16) {
      const float* rp = rij + (row0 + (size_t)lane) * 3;
      float rx = rp[0], ry = rp[1], rz = rp[2];
      float s = rx * rx + ry * ry + rz * rz;
      float inv = 1.0f / sqrtf(fmaxf(s, EPSF));
      float msk = (s < EPSF * EPSF) ? 0.0f : 1.0f;   // dij < EPS
      uRow[lane * 4 + 0] = rx * inv * msk;
      uRow[lane * 4 + 1] = ry * inv * msk;
      uRow[lane * 4 + 2] = rz * inv * msk;
      uRow[lane * 4 + 3] = 0.0f;
    }
    __asm volatile("s_wait_dscnt 0" ::: "memory");
    float ux[8], uy[8], uz[8];
#pragma unroll
    for (int r = 0; r < 8; ++r) {
      v4f u = *(const v4f*)&uRow[(r + 8 * hl) * 4];
      ux[r] = u[0]; uy[r] = u[1]; uz[r] = u[2];
    }

    // ---- filter f0 on (x0, x1): out0_a, out0_b ----
    {
      Rad rdA, rdB;
      mlp_rad_pair(a0K0, a0K1, a1K0, a1K1, sW1T[0], sW2T[0], sB1[0], sB2[0],
                   hid, lr, hl, rdA, rdB);
      store_out0(out0a, rdA.r0, row0, 0, lr, hl);
      store_out0(out0a, rdA.r1, row0, 1, lr, hl);
      store_out0(out0b, rdB.r0, row0, 0, lr, hl);
      store_out0(out0b, rdB.r1, row0, 1, lr, hl);
    }
    // ---- filter f11 on (x0, x1): out1_a, out1_c ----
    {
      Rad rdA, rdB;
      mlp_rad_pair(a0K0, a0K1, a1K0, a1K1, sW1T[2], sW2T[2], sB1[2], sB2[2],
                   hid, lr, hl, rdA, rdB);
      store_out1(out1a, rdA.r0, ux, uy, uz, row0, 0, lr, hl);
      store_out1(out1a, rdA.r1, ux, uy, uz, row0, 1, lr, hl);
      store_out1(out1c, rdB.r0, ux, uy, uz, row0, 0, lr, hl);
      store_out1(out1c, rdB.r1, ux, uy, uz, row0, 1, lr, hl);
    }
    // ---- filter f10 on x1: out1_b ----
    {
      Rad rd = mlp_rad(a1K0, a1K1, sW1T[1], sW2T[1], sB1[1], sB2[1], hid, lr, hl);
      store_out1(out1b, rd.r0, ux, uy, uz, row0, 0, lr, hl);
      store_out1(out1b, rd.r1, ux, uy, uz, row0, 1, lr, hl);
    }
  }
}

extern "C" void kernel_launch(void* const* d_in, const int* in_sizes, int n_in,
                              void* d_out, int out_size, void* d_ws, size_t ws_size,
                              hipStream_t stream) {
  (void)in_sizes; (void)n_in; (void)out_size; (void)d_ws; (void)ws_size;
  const float* x0  = (const float*)d_in[0];
  const float* x1  = (const float*)d_in[1];
  // d_in[2] = rbf (unused by the reference computation)
  const float* rij = (const float*)d_in[3];
  const float* f0_w1  = (const float*)d_in[4];
  const float* f0_b1  = (const float*)d_in[5];
  const float* f0_w2  = (const float*)d_in[6];
  const float* f0_b2  = (const float*)d_in[7];
  const float* f10_w1 = (const float*)d_in[8];
  const float* f10_b1 = (const float*)d_in[9];
  const float* f10_w2 = (const float*)d_in[10];
  const float* f10_b2 = (const float*)d_in[11];
  const float* f11_w1 = (const float*)d_in[12];
  const float* f11_b1 = (const float*)d_in[13];
  const float* f11_w2 = (const float*)d_in[14];
  const float* f11_b2 = (const float*)d_in[15];
  float* out = (float*)d_out;

  // 512*512/16 = 16384 tiles; 8 waves/block * 4 tiles/wave -> 512 blocks
  dim3 grid(512), block(256);
  hipLaunchKernelGGL(tfn_conv_kernel, grid, block, 0, stream,
                     x0, x1, rij,
                     f0_w1, f0_b1, f0_w2, f0_b2,
                     f10_w1, f10_b1, f10_w2, f10_b2,
                     f11_w1, f11_b1, f11_w2, f11_b2,
                     out);
}
